// Attention_68118181314928
// MI455X (gfx1250) — matrix operand order repaired and checked by reading, not compiled, not run on hardware
//
#include <hip/hip_runtime.h>
#include <hip/hip_bf16.h>

// Problem constants (match reference)
#define B_ 32
#define S_ 2048
#define E_ 1024
#define H_ 1024
#define LDSROW 1032   // bf16 elems per LDS row: 1024 + 8 pad -> 2064B row (16B aligned, bank-spread)

typedef __attribute__((ext_vector_type(16))) __bf16 v16bf;
typedef __attribute__((ext_vector_type(8)))  float  v8f;

struct BF16Frag { uint4 lo; uint4 hi; };   // 32B = 16 bf16

__device__ __forceinline__ unsigned pk_bf16(float a, float b) {
    __bf16 l = (__bf16)a, h = (__bf16)b;
    unsigned short ul = __builtin_bit_cast(unsigned short, l);
    unsigned short uh = __builtin_bit_cast(unsigned short, h);
    return (unsigned)ul | ((unsigned)uh << 16);
}

// Branch-free tanh using native trans ops (V_TANH_F32 if exposed, else exp2+rcp).
__device__ __forceinline__ float fast_tanh(float x) {
#if __has_builtin(__builtin_amdgcn_tanhf)
    return __builtin_amdgcn_tanhf(x);
#else
    // tanh(x) = 1 - 2/(exp(2x)+1); exp2 saturation gives exact +-1 tails.
    float t = __builtin_amdgcn_exp2f(x * 2.885390081777927f);   // exp(2x)
    return 1.f - 2.f * __builtin_amdgcn_rcpf(t + 1.f);
#endif
}

// ---------------- Kernel 1: W_k f32 -> bf16 ----------------
__global__ __launch_bounds__(256) void cvt_bf16_kernel(const float* __restrict__ in,
                                                       unsigned short* __restrict__ out) {
    int i = (blockIdx.x * 256 + threadIdx.x) * 4;          // H_*E_ = 1M elems
    float4 v = *(const float4*)(in + i);
    *(uint2*)(out + i) = make_uint2(pk_bf16(v.x, v.y), pk_bf16(v.z, v.w));
}

// ---------------- Kernel 2/7: GEMV out[b*H+h] = dot(W[h,:], X[b,:]) ----------------
__global__ __launch_bounds__(256) void gemv_kernel(const float* __restrict__ W,
                                                   const float* __restrict__ X,
                                                   float* __restrict__ out) {
    const int wid  = (blockIdx.x * 256 + threadIdx.x) >> 5;   // 0 .. B_*H_-1
    const int lane = threadIdx.x & 31;
    const int b = wid >> 10;            // / H_
    const int h = wid & (H_ - 1);
    const float* wr = W + (size_t)h * E_;
    const float* xr = X + (size_t)b * E_;
    float acc = 0.f;
    #pragma unroll
    for (int e = lane * 4; e < E_; e += 128) {
        float4 wv = *(const float4*)(wr + e);
        float4 xv = *(const float4*)(xr + e);
        acc += wv.x * xv.x + wv.y * xv.y + wv.z * xv.z + wv.w * xv.w;
    }
    for (int off = 16; off; off >>= 1) acc += __shfl_down(acc, off);
    if (lane == 0) out[wid] = acc;
}

// ---------------- Kernel 3: fused ctx GEMM + tanh + V-reduction -> score ----------------
__global__ __launch_bounds__(256) void attn_score_kernel(const float* __restrict__ enc,
                                                         const unsigned short* __restrict__ wkbf,
                                                         const float* __restrict__ wq,
                                                         const float* __restrict__ Vv,
                                                         float* __restrict__ score) {
    __shared__ unsigned short ldsEnc[32 * LDSROW];   // bf16 bits, ~64.5 KB
    __shared__ float ldsScore[32];

    const int tid = threadIdx.x;
    const int b   = blockIdx.y;
    const int s0  = blockIdx.x * 32;

    // Stage enc[b, s0:s0+32, :] -> bf16 LDS (coalesced f32 b128 loads + pack)
    const float* encTile = enc + ((size_t)b * S_ + s0) * E_;
    for (int i = tid; i < (32 * E_) / 4; i += 256) {
        int idx = i * 4;
        int s = idx >> 10;           // /E_
        int e = idx & (E_ - 1);
        float4 v = *(const float4*)(encTile + (size_t)s * E_ + e);
        *(uint2*)(&ldsEnc[s * LDSROW + e]) = make_uint2(pk_bf16(v.x, v.y), pk_bf16(v.z, v.w));
    }
    if (tid < 32) ldsScore[tid] = 0.f;
    __syncthreads();

    const int lane  = tid & 31;
    const int wave  = tid >> 5;
    const int lhalf = lane >> 4;     // 0 | 1
    const int l16   = lane & 15;
    const float* wqb = wq + (size_t)b * H_;

    // Each of 8 waves handles 8 h-tiles (16 rows each), both 16-wide s sub-tiles.
    #pragma unroll 1
    for (int mt = wave; mt < H_ / 16; mt += 8) {
        const int h0 = mt * 16;
        const unsigned short* wkRow = wkbf + (size_t)(h0 + l16) * E_;
        const int eA = lhalf * 8;    // A-frag K offset per ISA 16-bit A layout
        const int eB = lhalf * 8;    // B-frag K offset, same table order as A
        v8f acc0 = {}; v8f acc1 = {};

        // Keep unrolling modest: full unroll spills WMMA operands to scratch.
        #pragma unroll 2
        for (int k = 0; k < E_; k += 32) {
            BF16Frag af;
            af.lo = *(const uint4*)(wkRow + k + eA);        // K = eA .. eA+7
            af.hi = *(const uint4*)(wkRow + k + 16 + eA);   // K = eA+16 .. eA+23
            v16bf a = __builtin_bit_cast(v16bf, af);

            const unsigned short* bp0 = &ldsEnc[l16 * LDSROW + k + eB];
            const unsigned short* bp1 = &ldsEnc[(16 + l16) * LDSROW + k + eB];
            BF16Frag f0, f1;
            f0.lo = *(const uint4*)(bp0);     f0.hi = *(const uint4*)(bp0 + 16);
            f1.lo = *(const uint4*)(bp1);     f1.hi = *(const uint4*)(bp1 + 16);
            v16bf b0 = __builtin_bit_cast(v16bf, f0);
            v16bf b1 = __builtin_bit_cast(v16bf, f1);

            acc0 = __builtin_amdgcn_wmma_f32_16x16x32_bf16(false, a, false, b0,
                                                           (short)0, acc0, false, false);
            acc1 = __builtin_amdgcn_wmma_f32_16x16x32_bf16(false, a, false, b1,
                                                           (short)0, acc1, false, false);
        }

        // Epilogue: score[s] += sum_h V[h]*tanh(ctx+wq[h]).  acc elem r: M = r + 8*lhalf, N = l16.
        const int hb = h0 + lhalf * 8;   // multiple of 8 -> 32B aligned vector loads
        float wqv[8], vvv[8];
        *(float4*)(wqv)     = *(const float4*)(wqb + hb);
        *(float4*)(wqv + 4) = *(const float4*)(wqb + hb + 4);
        *(float4*)(vvv)     = *(const float4*)(Vv + hb);
        *(float4*)(vvv + 4) = *(const float4*)(Vv + hb + 4);
        float p0 = 0.f, p1 = 0.f;
        #pragma unroll
        for (int r = 0; r < 8; ++r) {
            p0 += fast_tanh(acc0[r] + wqv[r]) * vvv[r];
            p1 += fast_tanh(acc1[r] + wqv[r]) * vvv[r];
        }
        atomicAdd(&ldsScore[l16], p0);
        atomicAdd(&ldsScore[16 + l16], p1);
    }
    __syncthreads();
    if (tid < 32) score[(size_t)b * S_ + s0 + tid] = ldsScore[tid];
}

// ---------------- Kernel 4: softmax over S per batch -> alpha ----------------
__global__ __launch_bounds__(256) void softmax_kernel(const float* __restrict__ score,
                                                      float* __restrict__ alpha) {
    __shared__ float red[256];
    const int b = blockIdx.x, tid = threadIdx.x;
    const float* sr = score + (size_t)b * S_;
    float* ar = alpha + (size_t)b * S_;

    float m = -3.4e38f;
    for (int i = tid; i < S_; i += 256) m = fmaxf(m, sr[i]);
    red[tid] = m; __syncthreads();
    for (int o = 128; o; o >>= 1) { if (tid < o) red[tid] = fmaxf(red[tid], red[tid + o]); __syncthreads(); }
    m = red[0]; __syncthreads();

    float sum = 0.f;
    for (int i = tid; i < S_; i += 256) {
        float e = __builtin_amdgcn_exp2f((sr[i] - m) * 1.4426950408889634f);
        ar[i] = e; sum += e;
    }
    red[tid] = sum; __syncthreads();
    for (int o = 128; o; o >>= 1) { if (tid < o) red[tid] += red[tid + o]; __syncthreads(); }
    float inv = 1.f / red[0];
    for (int i = tid; i < S_; i += 256) ar[i] *= inv;
}

// ---------------- Kernel 5: pooled partials: part[b,sc,e] = sum_{s in chunk} alpha*enc ----------------
__global__ __launch_bounds__(256) void pool_partial_kernel(const float* __restrict__ enc,
                                                           const float* __restrict__ alpha,
                                                           float* __restrict__ part) {
    const int e  = blockIdx.x * 256 + threadIdx.x;   // grid.x = 4
    const int sc = blockIdx.y;                        // 16 chunks of 128 s
    const int b  = blockIdx.z;                        // 32
    const int s0 = sc * 128;
    const float* ep = enc + ((size_t)b * S_ + s0) * E_ + e;
    const float* ap = alpha + (size_t)b * S_ + s0;
    float acc = 0.f;
    for (int s = 0; s < 128; ++s) acc += ap[s] * ep[(size_t)s * E_];
    part[((size_t)b * 16 + sc) * E_ + e] = acc;
}

// ---------------- Kernel 6: reduce partials -> pooled[b,e] ----------------
__global__ __launch_bounds__(256) void pool_reduce_kernel(const float* __restrict__ part,
                                                          float* __restrict__ pooled) {
    const int i = blockIdx.x * 256 + threadIdx.x;    // B_*E_ = 32768
    const int b = i >> 10;
    const int e = i & (E_ - 1);
    const float* p = part + ((size_t)b * 16) * E_ + e;
    float acc = 0.f;
    #pragma unroll
    for (int sc = 0; sc < 16; ++sc) acc += p[(size_t)sc * E_];
    pooled[i] = acc;
}

extern "C" void kernel_launch(void* const* d_in, const int* in_sizes, int n_in,
                              void* d_out, int out_size, void* d_ws, size_t ws_size,
                              hipStream_t stream) {
    const float* enc = (const float*)d_in[0];   // [B,S,E]
    const float* dec = (const float*)d_in[1];   // [B,E]
    const float* Wk  = (const float*)d_in[2];   // [H,E]
    const float* Wq  = (const float*)d_in[3];   // [H,E]
    const float* Vv  = (const float*)d_in[4];   // [H]

    float* hidden = (float*)d_out;              // [B,H]
    float* score  = (float*)d_out + B_ * H_;    // [B,S]

    // Workspace layout (~4.5 MB)
    char* ws = (char*)d_ws;
    unsigned short* wkbf = (unsigned short*)ws;                 ws += (size_t)H_ * E_ * 2; // 2 MB
    float* wq     = (float*)ws;                                 ws += (size_t)B_ * H_ * 4; // 128 KB
    float* alpha  = (float*)ws;                                 ws += (size_t)B_ * S_ * 4; // 256 KB
    float* part   = (float*)ws;                                 ws += (size_t)B_ * 16 * E_ * 4; // 2 MB
    float* pooled = (float*)ws;                                                        // 128 KB

    // 1. W_k -> bf16
    cvt_bf16_kernel<<<(H_ * E_) / (256 * 4), 256, 0, stream>>>(Wk, wkbf);
    // 2. wq[b,h] = dec . W_q
    gemv_kernel<<<(B_ * H_ * 32) / 256, 256, 0, stream>>>(Wq, dec, wq);
    // 3. fused GEMM(tanh,V) -> score
    attn_score_kernel<<<dim3(S_ / 32, B_), 256, 0, stream>>>(enc, wkbf, wq, Vv, score);
    // 4. softmax -> alpha
    softmax_kernel<<<B_, 256, 0, stream>>>(score, alpha);
    // 5-6. pooled[b,e] = sum_s alpha*enc  (deterministic two-stage)
    pool_partial_kernel<<<dim3(E_ / 256, S_ / 128, B_), 256, 0, stream>>>(enc, alpha, part);
    pool_reduce_kernel<<<(B_ * E_) / 256, 256, 0, stream>>>(part, pooled);
    // 7. hidden[b,h] = pooled . W_k  (f32 weights)
    gemv_kernel<<<(B_ * H_ * 32) / 256, 256, 0, stream>>>(Wk, pooled, hidden);
}
